// SelfAttention_v1_46909632807066
// MI455X (gfx1250) — matrix operand order repaired and checked by reading, not compiled, not run on hardware
//
#include <hip/hip_runtime.h>

// ---------------------------------------------------------------------------
// Causal self-attention for MI455X (gfx1250), bf16 WMMA with f32 accumulate.
//   x:[4,2048,1024] f32, Wq/Wk/Wv:[1024,1024] f32  ->  out:[4,2048,1024] f32
// Pipeline: cast/transposes -> 3x proj GEMM -> QK^T GEMM (causal-skipped)
//           -> row softmax (writes bf16 P, zero tail) -> P*V GEMM -> d_out.
// Wave tile: 32x64 (2 A-fragments x 4 B-fragments -> 8 f32 accumulators),
// 8 waves/block -> 256x64 per block. All GEMM operands L2-resident (192 MB).
// ---------------------------------------------------------------------------

typedef __bf16 bf16;
typedef __attribute__((ext_vector_type(8)))  bf16  v8bf;
typedef __attribute__((ext_vector_type(16))) bf16  v16bf;
typedef __attribute__((ext_vector_type(8)))  float v8f;

static __device__ __forceinline__ v16bf cat8(v8bf lo, v8bf hi) {
  return __builtin_shufflevector(lo, hi, 0,1,2,3,4,5,6,7,8,9,10,11,12,13,14,15);
}

// A fragment: 16x32 bf16, rows m0..m0+15, contraction k0..k0+31.
// Lane L: row = m0+(L&15). Group g=L>>4: elements K = k0+16g .. +15 contiguous.
static __device__ __forceinline__ v16bf load_afrag(const bf16* __restrict__ A, long lda,
                                                   int m0, int k0, int lane) {
  int  row = m0 + (lane & 15);
  int  kb  = k0 + ((lane >> 4) << 4);
  const bf16* p = A + (long)row * lda + kb;
  v8bf lo = *(const v8bf*)p;         // K = kb .. kb+7
  v8bf hi = *(const v8bf*)(p + 8);   // K = kb+8 .. kb+15
  return cat8(lo, hi);
}

// B fragment: 32x16 bf16 from Bt (row-major B^T, ld = contraction length).
// Lane L: col n = n0+(L&15); group g: elements K = k0+16g .. +15 contiguous.
static __device__ __forceinline__ v16bf load_bfrag(const bf16* __restrict__ Bt, long ldb,
                                                   int n0, int k0, int lane) {
  int  col = n0 + (lane & 15);
  int  kb  = k0 + ((lane >> 4) << 4);
  const bf16* p = Bt + (long)col * ldb + kb;
  v8bf lo = *(const v8bf*)p;        // K = kb .. kb+7
  v8bf hi = *(const v8bf*)(p + 8);  // K = kb+8 .. kb+15
  return cat8(lo, hi);
}

static __device__ __forceinline__ v8f wmma_bf16(v16bf a, v16bf b, v8f c) {
  return __builtin_amdgcn_wmma_f32_16x16x32_bf16(false, a, false, b, (short)0, c,
                                                 false, false);
}

// --------------------------- elementwise helpers ---------------------------

__global__ __launch_bounds__(256) void k_cast_bf16(const float* __restrict__ in,
                                                   bf16* __restrict__ out, long n) {
  long i = (long)blockIdx.x * blockDim.x + threadIdx.x;
  if (i < n) out[i] = (bf16)in[i];
}

// Wt[o*D + i] = (bf16) W[i*D + o]
__global__ __launch_bounds__(256) void k_transpose_w(const float* __restrict__ W,
                                                     bf16* __restrict__ Wt, int D) {
  long idx = (long)blockIdx.x * blockDim.x + threadIdx.x;
  if (idx >= (long)D * D) return;
  int i = (int)(idx % D);
  int o = (int)(idx / D);
  Wt[(long)o * D + i] = (bf16)W[(long)i * D + o];
}

// Vt[b][e][s] = V[b][s][e]
__global__ __launch_bounds__(256) void k_transpose_v(const bf16* __restrict__ V,
                                                     bf16* __restrict__ Vt,
                                                     int S, int D, long n) {
  long idx = (long)blockIdx.x * blockDim.x + threadIdx.x;
  if (idx >= n) return;
  int  s = (int)(idx % S);
  int  e = (int)((idx / S) % D);
  long b = idx / ((long)S * D);
  Vt[idx] = V[b * (long)S * D + (long)s * D + e];
}

// ------------------------------- GEMMs -------------------------------------
// Wave w: rows m0 = bx*256 + 32w .. +31, cols n0 = by*64 .. +63.
// 8 accumulators: acc[t] rows m0..m0+15 of n-tile t; acc[4+t] rows m0+16..m0+31.

__global__ __launch_bounds__(256)
void k_proj_gemm(const bf16* __restrict__ A, const bf16* __restrict__ Bt,
                 bf16* __restrict__ C, int Kd, int N) {
  int lane = threadIdx.x & 31, wave = threadIdx.x >> 5;
  int m0 = blockIdx.x * 256 + wave * 32;
  int n0 = blockIdx.y * 64;
  v8f acc[8] = {};
  for (int k0 = 0; k0 < Kd; k0 += 32) {
    v16bf a0 = load_afrag(A, Kd, m0,      k0, lane);
    v16bf a1 = load_afrag(A, Kd, m0 + 16, k0, lane);
#pragma unroll
    for (int t = 0; t < 4; ++t) {
      v16bf b = load_bfrag(Bt, Kd, n0 + 16 * t, k0, lane);
      acc[t]     = wmma_bf16(a0, b, acc[t]);
      acc[4 + t] = wmma_bf16(a1, b, acc[4 + t]);
    }
  }
  int n = lane & 15, g = lane >> 4;
#pragma unroll
  for (int t = 0; t < 4; ++t)
#pragma unroll
    for (int v = 0; v < 8; ++v) {
      C[(long)(m0 + v + 8 * g) * N + (n0 + 16 * t + n)]      = (bf16)acc[t][v];
      C[(long)(m0 + 16 + v + 8 * g) * N + (n0 + 16 * t + n)] = (bf16)acc[4 + t][v];
    }
}

// scores[b] = Q[b] * K[b]^T  (row-major K IS B^T).  Causal: drop tiles n0 > m0+31.
__global__ __launch_bounds__(256)
void k_qk_gemm(const bf16* __restrict__ Q, const bf16* __restrict__ Kmat,
               float* __restrict__ Sc, int S, int Kd) {
  if (blockIdx.y * 64 > blockIdx.x * 256 + 255) return;  // whole block masked
  long b = blockIdx.z;
  const bf16* A  = Q    + b * (long)S * Kd;
  const bf16* Bt = Kmat + b * (long)S * Kd;
  float*      C  = Sc   + b * (long)S * S;
  int lane = threadIdx.x & 31, wave = threadIdx.x >> 5;
  int m0 = blockIdx.x * 256 + wave * 32;
  int n0 = blockIdx.y * 64;
  if (n0 > m0 + 31) return;                              // wave fully masked
  int ntiles = (m0 + 31 - n0) / 16 + 1;
  if (ntiles > 4) ntiles = 4;
  v8f acc[8] = {};
  for (int k0 = 0; k0 < Kd; k0 += 32) {
    v16bf a0 = load_afrag(A, Kd, m0,      k0, lane);
    v16bf a1 = load_afrag(A, Kd, m0 + 16, k0, lane);
#pragma unroll
    for (int t = 0; t < 4; ++t) {
      if (t < ntiles) {   // wave-uniform, EXEC stays all-ones
        v16bf bb = load_bfrag(Bt, Kd, n0 + 16 * t, k0, lane);
        acc[t]     = wmma_bf16(a0, bb, acc[t]);
        acc[4 + t] = wmma_bf16(a1, bb, acc[4 + t]);
      }
    }
  }
  int n = lane & 15, g = lane >> 4;
#pragma unroll
  for (int t = 0; t < 4; ++t)
    if (t < ntiles)
#pragma unroll
      for (int v = 0; v < 8; ++v) {
        C[(long)(m0 + v + 8 * g) * S + (n0 + 16 * t + n)]      = acc[t][v];
        C[(long)(m0 + 16 + v + 8 * g) * S + (n0 + 16 * t + n)] = acc[4 + t][v];
      }
}

// out[b] = P[b] * V[b].  Bt = Vt[b] ([D,S]).  K-loop bounded by causal extent.
__global__ __launch_bounds__(256)
void k_pv_gemm(const bf16* __restrict__ P, const bf16* __restrict__ Vt,
               float* __restrict__ Out, int S, int D) {
  long b = blockIdx.z;
  const bf16* A  = P   + b * (long)S * S;
  const bf16* Bt = Vt  + b * (long)D * S;
  float*      C  = Out + b * (long)S * D;
  int lane = threadIdx.x & 31, wave = threadIdx.x >> 5;
  int m0 = blockIdx.x * 256 + wave * 32;
  int n0 = blockIdx.y * 64;
  int kend = m0 + 32;              // cols <= m0+31 needed; m0 % 32 == 0
  if (kend > S) kend = S;
  v8f acc[8] = {};
  for (int k0 = 0; k0 < kend; k0 += 32) {
    v16bf a0 = load_afrag(A, S, m0,      k0, lane);
    v16bf a1 = load_afrag(A, S, m0 + 16, k0, lane);
#pragma unroll
    for (int t = 0; t < 4; ++t) {
      v16bf bb = load_bfrag(Bt, S, n0 + 16 * t, k0, lane);
      acc[t]     = wmma_bf16(a0, bb, acc[t]);
      acc[4 + t] = wmma_bf16(a1, bb, acc[4 + t]);
    }
  }
  int n = lane & 15, g = lane >> 4;
#pragma unroll
  for (int t = 0; t < 4; ++t)
#pragma unroll
    for (int v = 0; v < 8; ++v) {
      C[(long)(m0 + v + 8 * g) * D + (n0 + 16 * t + n)]      = acc[t][v];
      C[(long)(m0 + 16 + v + 8 * g) * D + (n0 + 16 * t + n)] = acc[4 + t][v];
    }
}

// ------------------------------ softmax ------------------------------------
// One block per row. Causal softmax of scale*scores, bf16 P out, zeros past col q.
__global__ __launch_bounds__(256)
void k_softmax_causal(const float* __restrict__ Sc, bf16* __restrict__ P,
                      int S, float scale) {
  long row = blockIdx.x;                 // b*S + q
  int  q   = (int)(row % S);
  const float* srow = Sc + row * S;
  bf16*        prow = P  + row * S;
  __shared__ float red[256];
  int tid = threadIdx.x;

  float m = -3.4e38f;
  for (int c = tid; c <= q; c += 256) m = fmaxf(m, srow[c]);
  red[tid] = m; __syncthreads();
  for (int off = 128; off > 0; off >>= 1) {
    if (tid < off) red[tid] = fmaxf(red[tid], red[tid + off]);
    __syncthreads();
  }
  m = red[0]; __syncthreads();

  float sum = 0.f;
  for (int c = tid; c <= q; c += 256) sum += __expf(scale * (srow[c] - m));
  red[tid] = sum; __syncthreads();
  for (int off = 128; off > 0; off >>= 1) {
    if (tid < off) red[tid] += red[tid + off];
    __syncthreads();
  }
  float inv = 1.0f / red[0];

  for (int c = tid; c < S; c += 256) {
    float w = (c <= q) ? __expf(scale * (srow[c] - m)) * inv : 0.0f;
    prow[c] = (bf16)w;
  }
}

// ------------------------------ launcher -----------------------------------

extern "C" void kernel_launch(void* const* d_in, const int* in_sizes, int n_in,
                              void* d_out, int out_size, void* d_ws, size_t ws_size,
                              hipStream_t stream) {
  (void)in_sizes; (void)n_in; (void)out_size; (void)ws_size;
  const int B = 4, S = 2048, D = 1024;
  const float* x  = (const float*)d_in[0];
  const float* Wq = (const float*)d_in[1];
  const float* Wk = (const float*)d_in[2];
  const float* Wv = (const float*)d_in[3];

  const long MB = 1l << 20;
  char* ws = (char*)d_ws;
  bf16*  xb  = (bf16*)(ws + 0);          // 16 MB
  bf16*  Wqt = (bf16*)(ws + 16 * MB);    //  2 MB
  bf16*  Wkt = (bf16*)(ws + 18 * MB);    //  2 MB
  bf16*  Wvt = (bf16*)(ws + 20 * MB);    //  2 MB
  bf16*  Qb  = (bf16*)(ws + 22 * MB);    // 16 MB
  bf16*  Kb  = (bf16*)(ws + 38 * MB);    // 16 MB
  bf16*  Vb  = (bf16*)(ws + 54 * MB);    // 16 MB
  bf16*  Vt  = (bf16*)(ws + 70 * MB);    // 16 MB
  float* Sc  = (float*)(ws + 86 * MB);   // 64 MB
  bf16*  P   = (bf16*)(ws + 22 * MB);    // 32 MB, overlays Q&K (dead after scores)

  const long nx = (long)B * S * D;       // 8M elements

  // 1) casts / weight transposes
  k_cast_bf16<<<(int)((nx + 255) / 256), 256, 0, stream>>>(x, xb, nx);
  int gw = (D * D + 255) / 256;
  k_transpose_w<<<gw, 256, 0, stream>>>(Wq, Wqt, D);
  k_transpose_w<<<gw, 256, 0, stream>>>(Wk, Wkt, D);
  k_transpose_w<<<gw, 256, 0, stream>>>(Wv, Wvt, D);

  // 2) Q/K/V projections: [8192,1024] x [1024,1024]
  dim3 gproj((B * S) / 256, D / 64);
  k_proj_gemm<<<gproj, 256, 0, stream>>>(xb, Wqt, Qb, D, D);
  k_proj_gemm<<<gproj, 256, 0, stream>>>(xb, Wkt, Kb, D, D);
  k_proj_gemm<<<gproj, 256, 0, stream>>>(xb, Wvt, Vb, D, D);

  // 3) V^T for the P*V B-fragments
  k_transpose_v<<<(int)((nx + 255) / 256), 256, 0, stream>>>(Vb, Vt, S, D, nx);

  // 4) scores = Q K^T (causal lower-triangle tiles only)
  dim3 gqk(S / 256, S / 64, B);
  k_qk_gemm<<<gqk, 256, 0, stream>>>(Qb, Kb, Sc, S, D);

  // 5) row softmax -> bf16 P (zero tail)
  k_softmax_causal<<<B * S, 256, 0, stream>>>(Sc, P, S, 1.0f / 32.0f);

  // 6) out = P V
  dim3 gpv(S / 256, D / 64, B);
  k_pv_gemm<<<gpv, 256, 0, stream>>>(P, Vt, (float*)d_out, S, D);
}
